// WoGCN_Encoder_55722905698998
// MI455X (gfx1250) — compile-verified, matrix operand order checked by reading, not run on hardware
//
#include <hip/hip_runtime.h>
#include <math.h>

// ---------------------------------------------------------------------------
// WoGCN encoder on MI455X (gfx1250, wave32, WMMA + double-buffered async LDS).
//   - bf16 WMMA (16x16x32) with f32 accumulation for both big GEMMs
//   - A tiles DMA'd via global_load_async_to_lds_b128, double buffered so the
//     DMA overlaps the WMMA burst (in-order ASYNCcnt => wait on <= PER)
//   - B fragments register-resident, shared across MT M-tiles (L2 BW saver)
//   - recurrence uses deterministic split-K=4 for per-step parallelism
// ---------------------------------------------------------------------------

typedef __bf16 bf16;
typedef __attribute__((ext_vector_type(16))) __bf16 v16bf;
typedef __attribute__((ext_vector_type(8)))  float  v8f;

#define B_SZ   32
#define T_SZ   48
#define S_SZ   30
#define IN_FTC 16
#define H1C    128
#define H2C    128
#define OUTC   64
#define GRU    3840      // H1*S = H2*S
#define G3     11520     // 3*GRU
#define KSPLIT 4
#define KCHUNK 320       // LDS staging chunk (cols of A), 40 x 16B segs/row

union FragBF {
  v16bf v;
  uint4 q[2];
};

// ---- issue the async DMA of one A chunk [MT*16 rows x 320 cols] into LDS ---
template <int MT>
__device__ __forceinline__ void stage_chunk(const bf16* __restrict__ A,
                                            int rowBase, int kOff, bf16* smBuf) {
  const int tid = threadIdx.x;
  constexpr int SEGS = KCHUNK / 8;                // 16B segments per row (40)
  constexpr int PER  = MT * 16 * SEGS / 256;      // transfers per thread
#pragma unroll
  for (int q = 0; q < PER; ++q) {
    int e   = q * 256 + tid;
    int row = e / SEGS;
    int seg = e - row * SEGS;
    const bf16* gp = A + (size_t)(rowBase + row) * GRU + kOff + seg * 8;
    unsigned lofs  = (unsigned)(size_t)(smBuf + row * KCHUNK + seg * 8);
    asm volatile("global_load_async_to_lds_b128 %0, %1, off"
                 :: "v"(lofs), "v"((unsigned long long)(size_t)gp)
                 : "memory");
  }
}

// ---------------------------------------------------------------------------
// WMMA K-loop: one wave computes MT 16x16 tiles sharing each B fragment.
// Fragment k mapping (bf16 16x16x32): element j of lane (l15|half) is
//   k = (j>>3)*16 + half*8 + (j&7)  -> two contiguous 16B chunks per lane.
// ---------------------------------------------------------------------------
template <int MT>
__device__ __forceinline__ void wmma_tiles(const bf16* __restrict__ A, int rowBase,
                                           const bf16* __restrict__ W, int gBase,
                                           int kBase, int kLen,
                                           bf16* sm /* [2*MT*16*KCHUNK] */,
                                           v8f acc[MT]) {
  const int lane = threadIdx.x & 31;
  const int l15  = lane & 15;
  const int half = lane >> 4;
  const bf16* bRow = W + (size_t)(gBase + l15) * GRU + kBase + half * 8;
  constexpr int SEGS  = KCHUNK / 8;
  constexpr int PER   = MT * 16 * SEGS / 256;
  constexpr int BUFSZ = MT * 16 * KCHUNK;

  stage_chunk<MT>(A, rowBase, kBase, sm);         // prologue: chunk 0 -> buf 0
  int p = 0;
  for (int kc = 0; kc < kLen; kc += KCHUNK, p ^= 1) {
    bool more = (kc + KCHUNK) < kLen;
    if (more) {                                   // overlap: DMA next chunk
      stage_chunk<MT>(A, rowBase, kBase + kc + KCHUNK, sm + (p ^ 1) * BUFSZ);
      // in-order completion: allow the newest PER (next chunk) to be in flight
      asm volatile("s_wait_asynccnt %0" :: "i"(PER) : "memory");
    } else {
      asm volatile("s_wait_asynccnt 0x0" ::: "memory");
    }
    __syncthreads();                              // current chunk visible to all

    const bf16* buf = sm + p * BUFSZ;
#pragma unroll
    for (int kk = 0; kk < KCHUNK; kk += 32) {
      FragBF fb;
      fb.q[0] = *(const uint4*)(bRow + kc + kk);
      fb.q[1] = *(const uint4*)(bRow + kc + kk + 16);
      __builtin_prefetch(bRow + kc + kk + KCHUNK, 0, 1);
      FragBF fa[MT];                              // independent A frags =>
#pragma unroll
      for (int m = 0; m < MT; ++m) {              //   one ds clause, one wait
        const bf16* lr = buf + (m * 16 + l15) * KCHUNK + kk + half * 8;
        fa[m].q[0] = *(const uint4*)(lr);
        fa[m].q[1] = *(const uint4*)(lr + 16);
      }
#pragma unroll
      for (int m = 0; m < MT; ++m) {              // back-to-back WMMAs
        acc[m] = __builtin_amdgcn_wmma_f32_16x16x32_bf16(
            /*neg_a=*/false, fa[m].v, /*neg_b=*/false, fb.v,
            /*c_mod=*/(short)0, acc[m], /*reuse_a=*/false, /*reuse_b=*/false);
      }
    }
    __syncthreads();                              // done reading buf p
  }
}

// --------------------------- weight fp32 -> bf16 ---------------------------
__global__ void cvt_weights(const float* __restrict__ Wih,
                            const float* __restrict__ Whh,
                            bf16* __restrict__ WihB, bf16* __restrict__ WhhB,
                            int n) {
  int i = blockIdx.x * 256 + threadIdx.x;
  if (i < n) {
    WihB[i] = (bf16)Wih[i];
    WhhB[i] = (bf16)Whh[i];
  }
}

// --------------------------- FC1 + ReLU -> xi (bf16) -----------------------
__global__ void fc1_relu(const float* __restrict__ x,
                         const float* __restrict__ fcw,
                         const float* __restrict__ fcb,
                         bf16* __restrict__ xi) {
  int idx = blockIdx.x * 256 + threadIdx.x;       // B*T*S*128 threads
  int c  = idx & 127;
  int s  = (idx >> 7) % S_SZ;
  int bt = idx / (128 * S_SZ);
  const float* xr = x + (size_t)(bt * S_SZ + s) * IN_FTC;
  const float* wr = fcw + c * IN_FTC;
  float acc = fcb[c];
#pragma unroll
  for (int i = 0; i < IN_FTC; ++i) acc += xr[i] * wr[i];
  acc = fmaxf(acc, 0.0f);
  xi[(size_t)bt * GRU + s * 128 + c] = (bf16)acc;
}

// --------------------- GEMM1: gx[t][b][g] = xi @ W_ih^T + b_ih -------------
// grid (90, 24): each block = 64 rows (4 M-tiles/wave) x 128 cols (8 waves)
__global__ void __launch_bounds__(256)
gemm_gx(const bf16* __restrict__ xi, const bf16* __restrict__ WihB,
        const float* __restrict__ b_ih, float* __restrict__ gx) {
  __shared__ bf16 sm[2 * 64 * KCHUNK];            // 80 KB double buffer
  const int wave    = threadIdx.x >> 5;
  const int gBase   = blockIdx.x * 128 + wave * 16;
  const int rowBase = blockIdx.y * 64;
  v8f acc[4] = {};
  wmma_tiles<4>(xi, rowBase, WihB, gBase, 0, GRU, sm, acc);

  const int lane = threadIdx.x & 31;
  const int l15  = lane & 15;
  const int half = lane >> 4;
  const int g    = gBase + l15;
  const float bias = b_ih[g];
#pragma unroll
  for (int m = 0; m < 4; ++m) {
#pragma unroll
    for (int r = 0; r < 8; ++r) {
      int bt = rowBase + m * 16 + r + 8 * half;   // D: vgpr r -> M=r (+8 hi)
      int t  = bt % T_SZ;
      int b  = bt / T_SZ;
      gx[(size_t)(t * B_SZ + b) * G3 + g] = acc[m][r] + bias;
    }
  }
}

// --------------- GEMM2 (per step): gh partials = h @ W_hh^T ----------------
// grid (90, KSPLIT): one B fragment feeds both M-tiles (full M=32);
// split-K gives 360 blocks/step; partials reduced deterministically later.
__global__ void __launch_bounds__(256)
gemm_gh(const bf16* __restrict__ hB, const bf16* __restrict__ WhhB,
        float* __restrict__ ghp) {
  __shared__ bf16 sm[2 * 32 * KCHUNK];            // 40 KB double buffer
  const int wave  = threadIdx.x >> 5;
  const int gBase = blockIdx.x * 128 + wave * 16;
  const int ks    = blockIdx.y;                   // 0..3
  v8f acc[2] = {};
  wmma_tiles<2>(hB, 0, WhhB, gBase, ks * (GRU / KSPLIT), GRU / KSPLIT, sm, acc);

  const int lane = threadIdx.x & 31;
  const int l15  = lane & 15;
  const int half = lane >> 4;
  const int g    = gBase + l15;
  float* dst = ghp + (size_t)ks * B_SZ * G3;
#pragma unroll
  for (int m = 0; m < 2; ++m) {
#pragma unroll
    for (int r = 0; r < 8; ++r) {
      int b = m * 16 + r + 8 * half;
      dst[(size_t)b * G3 + g] = acc[m][r];
    }
  }
}

// ----------------- GRU gate update (sums split-K partials) -----------------
__global__ void gru_gate(const float* __restrict__ gx_t,
                         const float* __restrict__ ghp,
                         const float* __restrict__ bhh,
                         float* __restrict__ h, bf16* __restrict__ hB) {
  int idx = blockIdx.x * 256 + threadIdx.x;       // B*GRU = 122880 threads
  int b = idx / GRU;
  int j = idx - b * GRU;
  const float* gxr = gx_t + (size_t)b * G3;
  const size_t base = (size_t)b * G3;
  float g0 = bhh[j], g1 = bhh[GRU + j], g2 = bhh[2 * GRU + j];
#pragma unroll
  for (int ks = 0; ks < KSPLIT; ++ks) {
    const float* p = ghp + (size_t)ks * B_SZ * G3 + base;
    g0 += p[j];
    g1 += p[GRU + j];
    g2 += p[2 * GRU + j];
  }
  float r = 1.0f / (1.0f + __expf(-(gxr[j] + g0)));
  float z = 1.0f / (1.0f + __expf(-(gxr[GRU + j] + g1)));
  float n = tanhf(gxr[2 * GRU + j] + r * g2);
  float hv = h[idx];
  float hn = (1.0f - z) * n + z * hv;
  h[idx]  = hn;
  hB[idx] = (bf16)hn;
}

// --------------------------- h zero init -----------------------------------
__global__ void init_h(float* __restrict__ h, bf16* __restrict__ hB) {
  int idx = blockIdx.x * 256 + threadIdx.x;
  h[idx]  = 0.0f;
  hB[idx] = (bf16)0.0f;
}

// --------------------------- FC2: out = relu(hT) @ fc2_w^T + fc2_b ---------
__global__ void fc2_out(const float* __restrict__ h,
                        const float* __restrict__ w2,
                        const float* __restrict__ b2,
                        float* __restrict__ out) {
  int idx = blockIdx.x * 256 + threadIdx.x;       // B*S*64 = 61440 threads
  int o = idx & 63;
  int s = (idx >> 6) % S_SZ;
  int b = idx / (64 * S_SZ);
  const float* hr = h + (size_t)b * GRU + s * 128;
  const float* wr = w2 + o * H2C;
  float acc = b2[o];
#pragma unroll 16
  for (int c = 0; c < H2C; ++c) acc += fmaxf(hr[c], 0.0f) * wr[c];
  out[idx] = acc;   // idx == (b*S + s)*64 + o
}

// ---------------------------------------------------------------------------
extern "C" void kernel_launch(void* const* d_in, const int* in_sizes, int n_in,
                              void* d_out, int out_size, void* d_ws, size_t ws_size,
                              hipStream_t stream) {
  const float* x   = (const float*)d_in[0];
  // d_in[1] = adj (unused, WoGCN ablation)
  const float* fcw = (const float*)d_in[2];
  const float* fcb = (const float*)d_in[3];
  const float* Wih = (const float*)d_in[4];
  const float* Whh = (const float*)d_in[5];
  const float* bih = (const float*)d_in[6];
  const float* bhh = (const float*)d_in[7];
  const float* w2  = (const float*)d_in[8];
  const float* b2  = (const float*)d_in[9];
  float* out = (float*)d_out;

  char* ws = (char*)d_ws;
  size_t off = 0;
  auto alloc = [&](size_t bytes) -> void* {
    void* p = ws + off;
    off += (bytes + 255) & ~(size_t)255;
    return p;
  };
  const size_t NW = (size_t)G3 * GRU;             // 44,236,800 elements
  bf16*  WihB = (bf16*)alloc(NW * sizeof(bf16));  // 88.5 MB
  bf16*  WhhB = (bf16*)alloc(NW * sizeof(bf16));  // 88.5 MB (L2-resident)
  bf16*  xiB  = (bf16*)alloc((size_t)B_SZ * T_SZ * GRU * sizeof(bf16));
  float* gx   = (float*)alloc((size_t)T_SZ * B_SZ * G3 * sizeof(float));
  float* ghp  = (float*)alloc((size_t)KSPLIT * B_SZ * G3 * sizeof(float));
  float* h    = (float*)alloc((size_t)B_SZ * GRU * sizeof(float));
  bf16*  hB   = (bf16*)alloc((size_t)B_SZ * GRU * sizeof(bf16));
  (void)ws_size; (void)in_sizes; (void)n_in; (void)out_size;

  // 1) weights -> bf16
  cvt_weights<<<(unsigned)((NW + 255) / 256), 256, 0, stream>>>(
      Wih, Whh, WihB, WhhB, (int)NW);

  // 2) FC1 + ReLU -> xi bf16
  fc1_relu<<<(B_SZ * T_SZ * S_SZ * H1C) / 256, 256, 0, stream>>>(x, fcw, fcb, xiB);

  // 3) gx = xi @ W_ih^T + b_ih (136 GFLOP, WMMA bf16, W_ih read 24x not 96x)
  gemm_gx<<<dim3(90, 24), 256, 0, stream>>>(xiB, WihB, bih, gx);

  // 4) h0 = 0
  init_h<<<(B_SZ * GRU) / 256, 256, 0, stream>>>(h, hB);

  // 5) recurrence: W_hh bf16 stays in L2; split-K=4 for per-step parallelism
  for (int t = 0; t < T_SZ; ++t) {
    gemm_gh<<<dim3(90, KSPLIT), 256, 0, stream>>>(hB, WhhB, ghp);
    gru_gate<<<(B_SZ * GRU) / 256, 256, 0, stream>>>(
        gx + (size_t)t * B_SZ * G3, ghp, bhh, h, hB);
  }

  // 6) out = relu(hT) @ fc2_w^T + fc2_b
  fc2_out<<<(B_SZ * S_SZ * OUTC) / 256, 256, 0, stream>>>(h, w2, b2, out);
}
